// AdaptiveGraphConstruction_5317169512655
// MI455X (gfx1250) — matrix operand order repaired and checked by reading, not compiled, not run on hardware
//
#include <hip/hip_runtime.h>
#include <hip/hip_bf16.h>

// Problem constants (from reference): N=128, C=64, T=256, V=25, IC=21
#define N_   128
#define C_   64
#define T_   256
#define V_   25
#define IC_  21
#define Tc   8                 // timesteps per chunk
#define COLS (Tc * V_)         // 200 columns per chunk (t-major, v fastest)
#define XP   72                // sX pitch (elems): 144B, 16B aligned, bank-skewed
#define EP   200               // sZq/sZk pitch (elems): 400B, 16B aligned
// Gram K layout: e = t_local*24 + o  (o padded 21->24) => K extent = 8*24 = 192 = 6 WMMA k-steps

typedef __attribute__((ext_vector_type(16))) __bf16 v16bf;
typedef __attribute__((ext_vector_type(8)))  __bf16 v8bf;
typedef __attribute__((ext_vector_type(4)))  __bf16 v4bf;
typedef __attribute__((ext_vector_type(8)))  float  v8f;

// A-fragment (16x32 bf16, row-major M x K in LDS):
// lane m = lane&15; lanes 0-15 hold K = kb+0..7 & kb+16..23; lanes 16-31 hold K = kb+8..15 & kb+24..31
__device__ __forceinline__ v16bf ldA(const __bf16* base, int pitch, int mbase, int kb, int lane) {
  const __bf16* p = base + (mbase + (lane & 15)) * pitch + kb + ((lane >> 4) << 3);
  v8bf lo = *(const v8bf*)p;          // 16B: K offsets +0..7   -> frag elems 0..7
  v8bf hi = *(const v8bf*)(p + 16);   // 16B: K offsets +16..23 -> frag elems 8..15
  return __builtin_shufflevector(lo, hi, 0,1,2,3,4,5,6,7,8,9,10,11,12,13,14,15);
}

// B-fragment (32x16 bf16). LDS holds B transposed: [n][K]. lane n = lane&15;
// lanes 0-15 hold K = kb+0..7 & kb+16..23; lanes 16-31 hold K = kb+8..15 & kb+24..31 (matches ldA).
__device__ __forceinline__ v16bf ldB(const __bf16* base, int pitch, int nbase, int kb, int lane) {
  const __bf16* p = base + (nbase + (lane & 15)) * pitch + kb + ((lane >> 4) << 3);
  v8bf lo = *(const v8bf*)p;          // 16B
  v8bf hi = *(const v8bf*)(p + 16);   // 16B: K offsets +16..23
  return __builtin_shufflevector(lo, hi, 0,1,2,3,4,5,6,7,8,9,10,11,12,13,14,15);
}

__device__ __forceinline__ v8f bwmma(v16bf a, v16bf b, v8f c) {
  return __builtin_amdgcn_wmma_f32_16x16x32_bf16(false, a, false, b, (short)0, c, false, false);
}

__global__ __launch_bounds__(256)
void agc_fused_kernel(const float* __restrict__ x,  const float* __restrict__ Wa,
                      const float* __restrict__ ba, const float* __restrict__ Wb,
                      const float* __restrict__ bb, const float* __restrict__ Aadj,
                      const float* __restrict__ alpha_p, float* __restrict__ out) {
  __shared__ __align__(16) __bf16 sW[64 * 64];       //  8,192 B: rows 0-20 Wa, 32-52 Wb, rest 0
  __shared__ __align__(16) float  sB[64];            //    256 B: biases (padded with 0)
  __shared__ __align__(16) __bf16 sX[208 * XP];      // 29,952 B: x chunk, [col][c] (transposed)
  __shared__ __align__(16) __bf16 sZq[32 * EP];      // 12,800 B: qa as [v][e], e = tl*24 + o
  __shared__ __align__(16) __bf16 sZk[32 * EP];      // 12,800 B: kb as [w][e]

  const int tid  = threadIdx.x;
  const int lane = tid & 31;
  const int wave = tid >> 5;
  const int n    = blockIdx.x;
  const float* __restrict__ xn = x + (size_t)n * (C_ * T_ * V_);

  // ---- one-time init: weights/bias to LDS (bf16), zero sZq/sZk ----
  for (int i = tid; i < 64 * 64; i += 256) {
    int r = i >> 6, c = i & 63;
    float w = 0.0f;
    if (r < IC_)                 w = Wa[r * C_ + c];
    else if (r >= 32 && r < 53)  w = Wb[(r - 32) * C_ + c];
    sW[i] = (__bf16)w;
  }
  if (tid < 64)
    sB[tid] = (tid < IC_) ? ba[tid] : (tid >= 32 && tid < 53) ? bb[tid - 32] : 0.0f;
  for (int i = tid; i < 32 * EP; i += 256) { sZq[i] = (__bf16)0.0f; sZk[i] = (__bf16)0.0f; }
  __syncthreads();

  // persistent Gram accumulator: 2x2 tiles of 16x16, one per wave pair (K split in half)
  const int btile = wave & 3;          // which 16x16 tile of GB
  const int kh2   = wave >> 2;         // which K half this wave accumulates
  const int vbB   = (btile >> 1) << 4; // GB tile row base
  const int wbB   = (btile & 1) << 4;  // GB tile col base
  v8f gb = {};

  // ================= main streaming loop over time chunks =================
  for (int t0 = 0; t0 < T_; t0 += Tc) {
    // ---- stage x[n, :, t0:t0+Tc, :] into LDS transposed to [col][c] (bf16) ----
    // each thread: 4x4 float patch (4 coalesced float4 loads) -> 4 packed b64 LDS stores
    {
      const float* __restrict__ xc = xn + t0 * V_;
      for (int i = tid; i < (C_ / 4) * (COLS / 4); i += 256) {   // 16*50 = 800 patches
        int cq   = i / (COLS / 4);
        int colq = i % (COLS / 4);
        int c = cq * 4, col = colq * 4;
        const float* p = xc + c * (T_ * V_) + col;
        float4 f0 = *(const float4*)(p);
        float4 f1 = *(const float4*)(p + 1 * (T_ * V_));
        float4 f2 = *(const float4*)(p + 2 * (T_ * V_));
        float4 f3 = *(const float4*)(p + 3 * (T_ * V_));
        v4bf q0 = { (__bf16)f0.x, (__bf16)f1.x, (__bf16)f2.x, (__bf16)f3.x };
        v4bf q1 = { (__bf16)f0.y, (__bf16)f1.y, (__bf16)f2.y, (__bf16)f3.y };
        v4bf q2 = { (__bf16)f0.z, (__bf16)f1.z, (__bf16)f2.z, (__bf16)f3.z };
        v4bf q3 = { (__bf16)f0.w, (__bf16)f1.w, (__bf16)f2.w, (__bf16)f3.w };
        *(v4bf*)&sX[(col + 0) * XP + c] = q0;
        *(v4bf*)&sX[(col + 1) * XP + c] = q1;
        *(v4bf*)&sX[(col + 2) * XP + c] = q2;
        *(v4bf*)&sX[(col + 3) * XP + c] = q3;
      }
      if (t0 + Tc < T_)  // hint next chunk into cache (global_prefetch_b8)
        __builtin_prefetch((const void*)(xn + (size_t)(tid & 63) * (T_ * V_) + (t0 + Tc) * V_), 0, 0);
    }
    __syncthreads();

    // ---- step A: Y[64 x 200] = W_cat @ X^T, packed scatter to sZq/sZk as [v][e] ----
    // 4 M-tiles x 13 N-tiles (cols padded 200->208), K = 64 = 2 WMMA steps
    for (int tile = wave; tile < 52; tile += 8) {
      int tm = tile & 3, tn = tile >> 2;
      v8f acc = {};
      acc = bwmma(ldA(sW, 64, tm << 4, 0,  lane), ldB(sX, XP, tn << 4, 0,  lane), acc);
      acc = bwmma(ldA(sW, 64, tm << 4, 32, lane), ldB(sX, XP, tn << 4, 32, lane), acc);
      int ncol = (tn << 4) + (lane & 15);
      if (ncol < COLS) {
        int tl = ncol / V_;            // t within chunk
        int v  = ncol - tl * V_;       // joint index
        int mb = (tm << 4) + ((lane >> 4) << 3);   // 8 consecutive output channels
        float4 b0 = *(const float4*)&sB[mb];
        float4 b1 = *(const float4*)&sB[mb + 4];
        v8bf pk;
        pk[0] = (__bf16)(acc[0] + b0.x); pk[1] = (__bf16)(acc[1] + b0.y);
        pk[2] = (__bf16)(acc[2] + b0.z); pk[3] = (__bf16)(acc[3] + b0.w);
        pk[4] = (__bf16)(acc[4] + b1.x); pk[5] = (__bf16)(acc[5] + b1.y);
        pk[6] = (__bf16)(acc[6] + b1.z); pk[7] = (__bf16)(acc[7] + b1.w);
        int grp = mb & 31;             // 0, 8, 16 valid (16 covers zero-pad o=21..23); 24 skipped
        if (grp != 24) {
          __bf16* dst = (mb < 32 ? sZq : sZk) + v * EP + tl * 24 + grp;
          *(v8bf*)dst = pk;            // single 16B ds_store_b128, aligned (48B*tl + 16B*grp/8)
        }
      }
    }
    __syncthreads();

    // ---- step B: GB[v,w] += sum_e Zq[v][e]*Zk[w][e], K = 192 ----
#pragma unroll
    for (int ks = 0; ks < 3; ++ks) {
      int kb2 = (kh2 * 3 + ks) * 32;
      gb = bwmma(ldA(sZq, EP, vbB, kb2, lane), ldB(sZk, EP, wbB, kb2, lane), gb);
    }
    // next iteration's post-stage barrier fences sZ reuse (WAR)
  }

  // ================= epilogue: reduce K halves, tanh/alpha/adjacency, softmax =================
  float* sRed = (float*)sX;            // reuse: 4 tiles x 256 floats = 4 KB
  {
    int mh = (lane >> 4) << 3;
    if (kh2 == 1) {
#pragma unroll
      for (int r = 0; r < 8; ++r)
        sRed[btile * 256 + (mh + r) * 16 + (lane & 15)] = gb[r];
    }
  }
  __syncthreads();

  float* sG = (float*)sZq;             // reuse: 32x32 floats = 4 KB
  if (kh2 == 0) {
    const float alpha = alpha_p[0];
    int mh = (lane >> 4) << 3;
    int w  = wbB + (lane & 15);
#pragma unroll
    for (int r = 0; r < 8; ++r) {
      int v = vbB + mh + r;
      float g = gb[r] + sRed[btile * 256 + (mh + r) * 16 + (lane & 15)];
      if (v < V_ && w < V_)
        sG[v * 32 + w] = Aadj[v * V_ + w] + tanhf(g * (1.0f / 5376.0f)) * alpha;
    }
  }
  __syncthreads();

  if (tid < V_) {
    int v = tid;
    float m = -1e30f;
#pragma unroll
    for (int w = 0; w < V_; ++w) m = fmaxf(m, sG[v * 32 + w]);
    float e[V_], s = 0.0f;
#pragma unroll
    for (int w = 0; w < V_; ++w) { e[w] = __expf(sG[v * 32 + w] - m); s += e[w]; }
    float inv = 1.0f / s;
    float* op = out + (size_t)n * (V_ * V_) + v * V_;
#pragma unroll
    for (int w = 0; w < V_; ++w) op[w] = e[w] * inv;
  }
}

extern "C" void kernel_launch(void* const* d_in, const int* in_sizes, int n_in,
                              void* d_out, int out_size, void* d_ws, size_t ws_size,
                              hipStream_t stream) {
  const float* x     = (const float*)d_in[0];
  const float* Wa    = (const float*)d_in[1];
  const float* ba    = (const float*)d_in[2];
  const float* Wb    = (const float*)d_in[3];
  const float* bb    = (const float*)d_in[4];
  const float* Aadj  = (const float*)d_in[5];
  const float* alpha = (const float*)d_in[6];
  agc_fused_kernel<<<dim3(N_), dim3(256), 0, stream>>>(
      x, Wa, ba, Wb, bb, Aadj, alpha, (float*)d_out);
}
